// SKUTBlock_61014305407372
// MI455X (gfx1250) — compile-verified, matrix operand order checked by reading, not run on hardware
//
#include <hip/hip_runtime.h>
#include <hip/hip_bf16.h>
#include <math.h>

#define ZB   64
#define NCTX 2048
#define DM   64
#define DFF  32
#define NEGV   (-1e9f)
#define EPSV   (1e-5f)
#define SM_SCALE 0.125f   // 1/sqrt(64)

typedef __attribute__((ext_vector_type(16))) _Float16 v16h;
typedef __attribute__((ext_vector_type(8)))  _Float16 v8h;
typedef __attribute__((ext_vector_type(8)))  float    v8f;

// ---- WMMA fragment helpers (CDNA5 16x16x32 f16 layouts, ISA 7.12.2) ----
// A (16x32, row-major source): lane holds row = lane&15; halves 0..7 cover
// K = k0 + hi*8 .. +7, halves 8..15 cover K = k0+16+hi*8 .. +7  (hi = lane>>4).
// B (32x16) consumes the K-major transpose with N = lane&15 -> identical loader.
__device__ __forceinline__ v16h frag_from_h16(const _Float16* rowp, int k0, int hi) {
    const v8h a = *(const v8h*)(rowp + k0 + hi * 8);
    const v8h b = *(const v8h*)(rowp + k0 + 16 + hi * 8);
    v16h r;
#pragma unroll
    for (int i = 0; i < 8; ++i) { r[i] = a[i]; r[i + 8] = b[i]; }
    return r;
}

__device__ __forceinline__ v16h frag_from_f32(const float* rowp, int k0, int hi) {
    v16h r;
#pragma unroll
    for (int i = 0; i < 8; ++i) {
        r[i]     = (_Float16)rowp[k0 + hi * 8 + i];
        r[i + 8] = (_Float16)rowp[k0 + 16 + hi * 8 + i];
    }
    return r;
}

__device__ __forceinline__ v8f wmma16(v16h a, v16h b, v8f c) {
    return __builtin_amdgcn_wmma_f32_16x16x32_f16(false, a, false, b, (short)0, c,
                                                  false, false);
}

__device__ __forceinline__ float hsum16(float v) {
    v += __shfl_xor(v, 1, 16);
    v += __shfl_xor(v, 2, 16);
    v += __shfl_xor(v, 4, 16);
    v += __shfl_xor(v, 8, 16);
    return v;
}
__device__ __forceinline__ float hmax16(float v) {
    v = fmaxf(v, __shfl_xor(v, 1, 16));
    v = fmaxf(v, __shfl_xor(v, 2, 16));
    v = fmaxf(v, __shfl_xor(v, 4, 16));
    v = fmaxf(v, __shfl_xor(v, 8, 16));
    return v;
}

// ---- async global -> LDS staging via inline asm (gfx1250 ASYNCcnt path) ----
// global_load_async_to_lds_b128 vdst(LDS byte offset), vaddr(64b global), off
typedef __attribute__((address_space(3))) _Float16 lds_f16;

__device__ __forceinline__ void async_copy16(const _Float16* gsrc, _Float16* ldst) {
    const unsigned loff = (unsigned)(size_t)(lds_f16*)ldst;  // flat -> LDS offset
    const unsigned long long ga = (unsigned long long)(size_t)gsrc;
    asm volatile("global_load_async_to_lds_b128 %0, %1, off"
                 :
                 : "v"(loff), "v"(ga)
                 : "memory");
}
__device__ __forceinline__ void wait_async_le2() {
    asm volatile("s_wait_asynccnt 0x2" ::: "memory");
}
__device__ __forceinline__ void wait_async_le0() {
    asm volatile("s_wait_asynccnt 0x0" ::: "memory");
}

// =====================================================================
// Kernel 1: Q = xWq, K = xWk, V = xWv  -> Qh,Kh row-major f16; Vt transposed f16
// =====================================================================
__global__ __launch_bounds__(256) void qkv_kernel(
    const float* __restrict__ x, const float* __restrict__ wq,
    const float* __restrict__ wk, const float* __restrict__ wv,
    _Float16* __restrict__ Qh, _Float16* __restrict__ Kh,
    _Float16* __restrict__ Vt) {
    __shared__ __align__(16) _Float16 sWt[3][DM * DM];  // transposed: [e][d]
    for (int idx = threadIdx.x; idx < DM * DM; idx += 256) {
        int d = idx / DM, e = idx % DM;
        sWt[0][e * DM + d] = (_Float16)wq[idx];
        sWt[1][e * DM + d] = (_Float16)wk[idx];
        sWt[2][e * DM + d] = (_Float16)wv[idx];
    }
    __syncthreads();

    const int wave = threadIdx.x >> 5, lane = threadIdx.x & 31;
    const int n16 = lane & 15, hi = (lane >> 4) & 1;
    const int z = blockIdx.y;
    const int m0 = blockIdx.x * 128 + wave * 16;

    const float* xrow = x + ((size_t)z * NCTX + m0 + n16) * DM;
    const v16h a0 = frag_from_f32(xrow, 0, hi);
    const v16h a1 = frag_from_f32(xrow, 32, hi);

#pragma unroll
    for (int mtx = 0; mtx < 3; ++mtx) {
#pragma unroll
        for (int t = 0; t < 4; ++t) {
            const _Float16* wrow = &sWt[mtx][(t * 16 + n16) * DM];
            v8f c = {};
            c = wmma16(a0, frag_from_h16(wrow, 0, hi), c);
            c = wmma16(a1, frag_from_h16(wrow, 32, hi), c);
            if (mtx == 2) {  // V stored transposed: Vt[z][d][token]
                const int d = t * 16 + n16;
                v8h vs;
#pragma unroll
                for (int r = 0; r < 8; ++r) vs[r] = (_Float16)c[r];
                *(v8h*)(Vt + ((size_t)z * DM + d) * NCTX + m0 + hi * 8) = vs;
            } else {
                _Float16* dst = (mtx == 0 ? Qh : Kh) +
                                ((size_t)z * NCTX + m0 + hi * 8) * DM + t * 16 + n16;
#pragma unroll
                for (int r = 0; r < 8; ++r) dst[r * DM] = (_Float16)c[r];
            }
        }
    }
}

// =====================================================================
// Kernel 2: causal flash attention + residual + LayerNorm -> y (f32)
// K/V tiles async-staged to LDS, double buffered, shared by all 8 waves.
// =====================================================================
__global__ __launch_bounds__(256) void attn_kernel(
    const float* __restrict__ x, const _Float16* __restrict__ Qh,
    const _Float16* __restrict__ Kh, const _Float16* __restrict__ Vt,
    const float* __restrict__ kpm, float* __restrict__ y) {
    __shared__ __align__(16) _Float16 sK[2][32 * DM];   // [key][d]   4KB each
    __shared__ __align__(16) _Float16 sV[2][DM * 32];   // [d][key]   4KB each
    __shared__ __align__(16) _Float16 sP[8][16 * 32];   // per-wave P transpose

    const int tid = threadIdx.x;
    const int wave = tid >> 5, lane = tid & 31;
    const int n16 = lane & 15, hi = (lane >> 4) & 1;
    const int z = blockIdx.y;
    const int m0b = blockIdx.x * 128;
    const int m0 = m0b + wave * 16;
    const int nt = (m0b + 159) >> 5;  // 32-key tiles covering keys 0..m0b+127

    // cooperative copy mapping: one 16B chunk per thread per matrix
    const int krow = tid >> 3, kch = tid & 7;  // K: 32 rows x 8 chunks
    const int vrow = tid >> 2, vch = tid & 3;  // Vt: 64 rows x 4 chunks
    const _Float16* Kz = Kh + (size_t)z * NCTX * DM;
    const _Float16* Vz = Vt + (size_t)z * DM * NCTX;

    const _Float16* qrow = Qh + ((size_t)z * NCTX + m0 + n16) * DM;
    const v16h qa0 = frag_from_h16(qrow, 0, hi);
    const v16h qa1 = frag_from_h16(qrow, 32, hi);

    v8f oacc[4] = {{}, {}, {}, {}};
    float mrow[8], lrow[8];
#pragma unroll
    for (int r = 0; r < 8; ++r) { mrow[r] = -1e30f; lrow[r] = 0.f; }

    _Float16* pw = sP[wave];

    // prologue: stage tile 0
    {
        async_copy16(Kz + (size_t)krow * DM + kch * 8, &sK[0][krow * DM + kch * 8]);
        async_copy16(Vz + (size_t)vrow * NCTX + vch * 8, &sV[0][vrow * 32 + vch * 8]);
    }

    for (int j = 0; j < nt; ++j) {
        const int cur = j & 1;
        if (j + 1 < nt) {  // prefetch next tile into the other buffer
            const int kb = (j + 1) * 32;
            async_copy16(Kz + (size_t)(kb + krow) * DM + kch * 8,
                         &sK[cur ^ 1][krow * DM + kch * 8]);
            async_copy16(Vz + (size_t)vrow * NCTX + kb + vch * 8,
                         &sV[cur ^ 1][vrow * 32 + vch * 8]);
            wait_async_le2();
        } else {
            wait_async_le0();
        }
        __syncthreads();  // tile j visible to all waves

        const int kbase = j * 32;
        if (kbase <= m0 + 15) {  // wave-uniform causal skip (EXEC stays all-ones)
            // ---- scores: S = Q * K^T from LDS ----
            const _Float16* krow0 = &sK[cur][n16 * DM];
            const _Float16* krow1 = krow0 + 16 * DM;
            v8f s0 = {}, s1 = {};
            s0 = wmma16(qa0, frag_from_h16(krow0, 0, hi), s0);
            s0 = wmma16(qa1, frag_from_h16(krow0, 32, hi), s0);
            s1 = wmma16(qa0, frag_from_h16(krow1, 0, hi), s1);
            s1 = wmma16(qa1, frag_from_h16(krow1, 32, hi), s1);

            const float kp0 = kpm[(size_t)z * NCTX + kbase + n16];
            const float kp1 = kpm[(size_t)z * NCTX + kbase + 16 + n16];
            const int col0 = kbase + n16, col1 = col0 + 16;

            // ---- mask + scale + online softmax ----
#pragma unroll
            for (int r = 0; r < 8; ++r) {
                const int row = m0 + r + 8 * hi;
                float v0 = (s0[r] + kp0 + (col0 > row ? NEGV : 0.f)) * SM_SCALE;
                float v1 = (s1[r] + kp1 + (col1 > row ? NEGV : 0.f)) * SM_SCALE;
                const float tmax = hmax16(fmaxf(v0, v1));
                const float mn = fmaxf(mrow[r], tmax);
                const float corr = __expf(mrow[r] - mn);
                mrow[r] = mn;
                const float p0 = __expf(v0 - mn);
                const float p1 = __expf(v1 - mn);
                s0[r] = p0; s1[r] = p1;
                lrow[r] = lrow[r] * corr + hsum16(p0 + p1);
                oacc[0][r] *= corr; oacc[1][r] *= corr;
                oacc[2][r] *= corr; oacc[3][r] *= corr;
            }

            // ---- transpose P (C-layout f32) -> A-layout f16 through LDS ----
#pragma unroll
            for (int r = 0; r < 8; ++r) {
                pw[(r + 8 * hi) * 32 + n16]      = (_Float16)s0[r];
                pw[(r + 8 * hi) * 32 + 16 + n16] = (_Float16)s1[r];
            }
            asm volatile("s_wait_dscnt 0" ::: "memory");
            const v16h pa = frag_from_h16(pw + n16 * 32, 0, hi);

            // ---- O += P * V from LDS ----
#pragma unroll
            for (int t = 0; t < 4; ++t) {
                const _Float16* vrow2 = &sV[cur][(t * 16 + n16) * 32];
                oacc[t] = wmma16(pa, frag_from_h16(vrow2, 0, hi), oacc[t]);
            }
        }
        __syncthreads();  // all waves done with buffer `cur` before it is refilled
    }

    // ---- epilogue: 1/l, residual, LayerNorm over D=64, store y ----
    float vals[4][8];
#pragma unroll
    for (int r = 0; r < 8; ++r) {
        const float inv = 1.f / lrow[r];
        const size_t rb = ((size_t)z * NCTX + m0 + r + 8 * hi) * DM;
#pragma unroll
        for (int t = 0; t < 4; ++t)
            vals[t][r] = oacc[t][r] * inv + x[rb + t * 16 + n16];
    }
#pragma unroll
    for (int r = 0; r < 8; ++r) {
        float s = vals[0][r] + vals[1][r] + vals[2][r] + vals[3][r];
        const float mu = hsum16(s) * (1.f / DM);
        float vv = 0.f;
#pragma unroll
        for (int t = 0; t < 4; ++t) {
            const float d = vals[t][r] - mu;
            vv += d * d;
        }
        const float rs = rsqrtf(hsum16(vv) * (1.f / DM) + EPSV);
        const size_t rb = ((size_t)z * NCTX + m0 + r + 8 * hi) * DM;
#pragma unroll
        for (int t = 0; t < 4; ++t)
            y[rb + t * 16 + n16] = (vals[t][r] - mu) * rs;
    }
}

// =====================================================================
// Kernel 3: FFN (relu(y@w1)@w2) + residual + LayerNorm -> out
// =====================================================================
__global__ __launch_bounds__(256) void ffn_kernel(
    const float* __restrict__ y, const float* __restrict__ w1,
    const float* __restrict__ w2, float* __restrict__ out) {
    __shared__ __align__(16) _Float16 sW1t[DFF * DM];  // [f][d] = w1[d][f]
    __shared__ __align__(16) _Float16 sW2t[DM * DFF];  // [d][f] = w2[f][d]
    __shared__ __align__(16) _Float16 sH[8][16 * 32];

    for (int idx = threadIdx.x; idx < DM * DFF; idx += 256) {
        int d = idx / DFF, f = idx % DFF;
        sW1t[f * DM + d] = (_Float16)w1[idx];
    }
    for (int idx = threadIdx.x; idx < DFF * DM; idx += 256) {
        int f = idx / DM, d = idx % DM;
        sW2t[d * DFF + f] = (_Float16)w2[idx];
    }
    __syncthreads();

    const int wave = threadIdx.x >> 5, lane = threadIdx.x & 31;
    const int n16 = lane & 15, hi = (lane >> 4) & 1;
    const int z = blockIdx.y;
    const int m0 = blockIdx.x * 128 + wave * 16;

    const float* yrow = y + ((size_t)z * NCTX + m0 + n16) * DM;
    const v16h a0 = frag_from_f32(yrow, 0, hi);
    const v16h a1 = frag_from_f32(yrow, 32, hi);

    // h = relu(y @ w1): 16x32 result (two 16-wide tiles)
    _Float16* pw = sH[wave];
#pragma unroll
    for (int t = 0; t < 2; ++t) {
        const _Float16* wrow = sW1t + (t * 16 + n16) * DM;
        v8f h = {};
        h = wmma16(a0, frag_from_h16(wrow, 0, hi), h);
        h = wmma16(a1, frag_from_h16(wrow, 32, hi), h);
#pragma unroll
        for (int r = 0; r < 8; ++r)
            pw[(r + 8 * hi) * 32 + t * 16 + n16] = (_Float16)fmaxf(h[r], 0.f);
    }
    asm volatile("s_wait_dscnt 0" ::: "memory");
    const v16h ha = frag_from_h16(pw + n16 * 32, 0, hi);

    // ff = h @ w2 (K = 32), then residual + LN
    float vals[4][8];
#pragma unroll
    for (int t = 0; t < 4; ++t) {
        const _Float16* wrow = sW2t + (t * 16 + n16) * DFF;
        v8f f = {};
        f = wmma16(ha, frag_from_h16(wrow, 0, hi), f);
#pragma unroll
        for (int r = 0; r < 8; ++r) {
            const size_t rb = ((size_t)z * NCTX + m0 + r + 8 * hi) * DM;
            vals[t][r] = f[r] + y[rb + t * 16 + n16];
        }
    }
#pragma unroll
    for (int r = 0; r < 8; ++r) {
        float s = vals[0][r] + vals[1][r] + vals[2][r] + vals[3][r];
        const float mu = hsum16(s) * (1.f / DM);
        float vv = 0.f;
#pragma unroll
        for (int t = 0; t < 4; ++t) {
            const float d = vals[t][r] - mu;
            vv += d * d;
        }
        const float rs = rsqrtf(hsum16(vv) * (1.f / DM) + EPSV);
        const size_t rb = ((size_t)z * NCTX + m0 + r + 8 * hi) * DM;
#pragma unroll
        for (int t = 0; t < 4; ++t)
            out[rb + t * 16 + n16] = (vals[t][r] - mu) * rs;
    }
}

// =====================================================================
extern "C" void kernel_launch(void* const* d_in, const int* in_sizes, int n_in,
                              void* d_out, int out_size, void* d_ws, size_t ws_size,
                              hipStream_t stream) {
    const float* x   = (const float*)d_in[0];
    const float* wq  = (const float*)d_in[1];
    const float* wk  = (const float*)d_in[2];
    const float* wv  = (const float*)d_in[3];
    const float* w1  = (const float*)d_in[4];
    const float* w2  = (const float*)d_in[5];
    const float* kpm = (const float*)d_in[6];
    // d_in[7] = causal mask, reproduced analytically in-kernel
    float* out = (float*)d_out;

    const size_t tokElems = (size_t)ZB * NCTX * DM;
    _Float16* Qh = (_Float16*)d_ws;
    _Float16* Kh = Qh + tokElems;
    _Float16* Vt = Kh + tokElems;
    float*    y  = (float*)(Vt + tokElems);

    dim3 grid(NCTX / 128, ZB);
    qkv_kernel<<<grid, 256, 0, stream>>>(x, wq, wk, wv, Qh, Kh, Vt);
    attn_kernel<<<grid, 256, 0, stream>>>(x, Qh, Kh, Vt, kpm, y);
    ffn_kernel<<<grid, 256, 0, stream>>>(y, w1, w2, out);
}
